// VLSTM_32882269618464
// MI455X (gfx1250) — matrix operand order repaired and checked by reading, not compiled, not run on hardware
//
#include <hip/hip_runtime.h>
#include <hip/hip_bf16.h>
#include <cstdint>

// ---------------------------------------------------------------------------
// VLSTM for MI455X (gfx1250, wave32, WMMA + Tensor Data Mover).
//   B=32, T=2048, IX=IH=H=256.
//
// Pass 0a: x, h_prev fp32 -> bf16 row-major in d_ws            (memory-bound)
// Pass 0b: 9 weight matrices fp32 [K][N] -> bf16 [N][K] in d_ws
// Pass 1 : pre_g = x@Wx_g + h@Wh_g + b via bf16 WMMA 16x16x32 (f32 acc).
//          A tiles staged to LDS by the TDM (tensor_load_to_lds), double-
//          buffered so DMA of chunk c+1 overlaps WMMA on chunk c. A-fragment
//          ds loads are software-pipelined one deep past the WMMAs.
// Pass 2 : sequential scan, one persistent 1024-thread workgroup (32 waves);
//          u_i lives in LDS (loaded once by one TDM op), h exchanged through
//          LDS as bf16, c in registers, pres loads hoisted over the WMMAs.
// ---------------------------------------------------------------------------

typedef __attribute__((ext_vector_type(16))) __bf16 v16bf;
typedef __attribute__((ext_vector_type(8)))  __bf16 v8bf;
typedef __attribute__((ext_vector_type(8)))  float  v8f;
typedef __attribute__((ext_vector_type(4)))  unsigned int u32x4;
typedef __attribute__((ext_vector_type(8)))  int    i32x8;
typedef __attribute__((ext_vector_type(4)))  int    i32x4;

#define BB   32
#define TT   2048
#define HH   256
#define BT   (BB * TT)                 // 65536 rows
#define GSZ  ((size_t)BT * HH)         // elements per gate pre-tensor
#define WSZ  ((size_t)HH * HH)         // one 256x256 matrix

static __device__ __forceinline__ float sigm(float x) {
  return 1.0f / (1.0f + __expf(-x));
}

static __device__ __forceinline__ v16bf cat16(v8bf lo, v8bf hi) {
  return __builtin_shufflevector(lo, hi, 0, 1, 2, 3, 4, 5, 6, 7,
                                         8, 9, 10, 11, 12, 13, 14, 15);
}

// ---------------------------------------------------------------------------
// TDM: DMA a 2-D tile of bf16 (tile_x elems per row, tile_y rows, row stride
// in elements) from global memory into LDS. D# per CDNA5 ISA §8.3-8.5.
// 6-arg builtin: (u32x4 g0, i32x8 g1, i32x4 g2, i32x4 g3, i32x8 g4, i32 cpol)
// Issue from ONE wave only; complete with s_wait_tensorcnt(0) + barrier.
// ---------------------------------------------------------------------------
static __device__ __forceinline__ void tdm_load_2d_bf16(
    unsigned int lds_addr, const void* gptr,
    unsigned int tile_x, unsigned int tile_y,
    unsigned int stride_elems, unsigned int tensor_rows) {
  const unsigned long long ga = (unsigned long long)(uintptr_t)gptr;
  u32x4 g0;
  g0[0] = 1u;                                   // count=1 (valid descriptor)
  g0[1] = lds_addr;                             // LDS byte address
  g0[2] = (unsigned int)ga;                     // global_addr[31:0]
  g0[3] = (unsigned int)((ga >> 32) & 0x01FFFFFFu) | (2u << 30);  // +type=2
  i32x8 g1;
  g1[0] = (int)(1u << 16);                      // data_size=1 (2 bytes)
  g1[1] = (int)((stride_elems & 0xFFFFu) << 16);            // tensor_dim0 lo
  g1[2] = (int)(((stride_elems >> 16) & 0xFFFFu) |
                ((tensor_rows & 0xFFFFu) << 16));           // dim0 hi|dim1 lo
  g1[3] = (int)(((tensor_rows >> 16) & 0xFFFFu) |
                ((tile_x & 0xFFFFu) << 16));                // dim1 hi|tile_d0
  g1[4] = (int)(tile_y & 0xFFFFu);                          // tile_dim1
  g1[5] = (int)stride_elems;                                // dim0_stride lo32
  g1[6] = 0;                                                // stride hi16
  g1[7] = 0;
  const i32x4 gz4 = {0, 0, 0, 0};               // 2-D tensor: groups 2,3 unused
  const i32x8 gz8 = {0, 0, 0, 0, 0, 0, 0, 0};
  __builtin_amdgcn_tensor_load_to_lds(g0, g1, gz4, gz4, gz8, 0);
}

static __device__ __forceinline__ unsigned int lds_off(const void* p) {
  // Generic pointer to LDS: low 32 bits are the LDS byte address (aperture).
  return (unsigned int)(uintptr_t)p;
}

// ---------------------------------------------------------------------------
// Pass 0a: fp32 -> bf16, row-major, 8 elements per thread.
// ---------------------------------------------------------------------------
__global__ __launch_bounds__(256)
void vlstm_cvt_bf16(const float* __restrict__ in, __bf16* __restrict__ out) {
  const size_t i = ((size_t)blockIdx.x * blockDim.x + threadIdx.x) * 8;
  const float4 a = *(const float4*)(in + i);
  const float4 b = *(const float4*)(in + i + 4);
  v8bf o;
  o[0] = (__bf16)a.x; o[1] = (__bf16)a.y; o[2] = (__bf16)a.z; o[3] = (__bf16)a.w;
  o[4] = (__bf16)b.x; o[5] = (__bf16)b.y; o[6] = (__bf16)b.z; o[7] = (__bf16)b.w;
  *(v8bf*)(out + i) = o;
}

// ---------------------------------------------------------------------------
// Pass 0b: convert + transpose 9 weight matrices [K][N] fp32 -> [N][K] bf16.
// ---------------------------------------------------------------------------
__global__ __launch_bounds__(256)
void vlstm_cvt_wT(const float* __restrict__ w0, const float* __restrict__ w1,
                  const float* __restrict__ w2, const float* __restrict__ w3,
                  const float* __restrict__ w4, const float* __restrict__ w5,
                  const float* __restrict__ w6, const float* __restrict__ w7,
                  const float* __restrict__ w8, __bf16* __restrict__ wT) {
  const int z = blockIdx.z;
  const float* src =
      (z == 0) ? w0 : (z == 1) ? w1 : (z == 2) ? w2 : (z == 3) ? w3 :
      (z == 4) ? w4 : (z == 5) ? w5 : (z == 6) ? w6 : (z == 7) ? w7 : w8;
  const int idx = blockIdx.x * 256 + threadIdx.x;   // 0..65535
  const int k = idx >> 8;
  const int n = idx & 255;
  wT[(size_t)z * WSZ + (size_t)n * HH + k] = (__bf16)src[(size_t)k * HH + n];
}

// ---------------------------------------------------------------------------
// Pass 1: pre-activation GEMM. grid = (BT/256, 2 N-halves, 4 gates),
// block = 256 threads (8 waves). Wave: one 16-col N tile x 16 M tiles.
// A staged via TDM into LDS (2 x 16KB double buffer, 16 chunks:
// 8 K-steps of x then 8 of h_prev). B frags from global (1 MB, L2-hot).
// ---------------------------------------------------------------------------
__global__ __launch_bounds__(256)
void vlstm_pre_gemm(const __bf16* __restrict__ xbf,
                    const __bf16* __restrict__ hbf,
                    const __bf16* __restrict__ wT,   // 8 matrices [N][K]
                    const float* __restrict__ bias,
                    float* __restrict__ pres) {
  __shared__ __bf16 tileA[2][256 * 32];   // [buf][row*32 + k], 16 KB each

  const int lane  = threadIdx.x & 31;
  const int wv    = threadIdx.x >> 5;          // 0..7
  const int gate  = blockIdx.z;                // 0..3
  const int nTile = blockIdx.y * 8 + wv;       // 0..15
  const int ncol0 = nTile * 16;
  const int m0    = blockIdx.x * 256;          // M block base row

  const int l15   = lane & 15;
  const int lhalf = lane >> 4;

  v8f zero = {};
  v8f acc[16];
#pragma unroll
  for (int i = 0; i < 16; ++i) acc[i] = zero;

  // chunk c: pass = c>>3 (0 = x, 1 = h_prev), kk = (c&7)*32
  if (wv == 0) {
    tdm_load_2d_bf16(lds_off(&tileA[0][0]), xbf + (size_t)m0 * HH,
                     /*tile_x=*/32, /*tile_y=*/256,
                     /*stride=*/HH, /*rows=*/256);
  }

  for (int c = 0; c < 16; ++c) {
    if (wv == 0) __builtin_amdgcn_s_wait_tensorcnt(0);  // chunk c in LDS
    __syncthreads();                                    // publish to all waves
    if (wv == 0 && c + 1 < 16) {
      const int pass1 = (c + 1) >> 3;
      const int kk1   = ((c + 1) & 7) * 32;
      const __bf16* src = (pass1 ? hbf : xbf) + (size_t)m0 * HH + kk1;
      tdm_load_2d_bf16(lds_off(&tileA[(c + 1) & 1][0]), src,
                       32, 256, HH, 256);               // overlaps compute
    }

    const int pass = c >> 3;
    const int kk   = (c & 7) * 32;
    // B fragment: K {8h..8h+7, 16+8h..16+8h+7} for this lane's N column.
    const __bf16* W = wT + (size_t)(gate * 2 + pass) * WSZ
                         + (size_t)(ncol0 + l15) * HH;
    const __bf16* wb = W + kk + (lhalf ? 8 : 0);
    const v16bf bfrag = cat16(*(const v8bf*)wb, *(const v8bf*)(wb + 16));

    const __bf16* tb = &tileA[c & 1][0] + (lhalf ? 8 : 0);

    // Software-pipeline the A-fragment LDS loads one tile deep so the next
    // ds_load pair issues before the wait that feeds the current WMMA.
    v16bf afrag;
    {
      const __bf16* ap = tb + (size_t)l15 * 32;
      afrag = cat16(*(const v8bf*)ap, *(const v8bf*)(ap + 16));
    }
#pragma unroll
    for (int mt = 0; mt < 16; ++mt) {
      v16bf anext;
      if (mt + 1 < 16) {
        const __bf16* ap = tb + (size_t)((mt + 1) * 16 + l15) * 32;
        anext = cat16(*(const v8bf*)ap, *(const v8bf*)(ap + 16));
      }
      acc[mt] = __builtin_amdgcn_wmma_f32_16x16x32_bf16(
          false, afrag, false, bfrag, (short)0, acc[mt], false, false);
      afrag = anext;
    }
    // Buffer (c&1) is reused by the TDM issued in iteration c+1 (chunk c+2), which is
    // after iteration c+1's top barrier -> no extra barrier needed here.
  }

  // Epilogue: add bias (depends on N only), store fp32.
  const float bv = bias[ncol0 + l15];
  float* outg = pres + (size_t)gate * GSZ;
#pragma unroll
  for (int mt = 0; mt < 16; ++mt) {
#pragma unroll
    for (int v = 0; v < 8; ++v) {
      const int m = m0 + mt * 16 + v + (lhalf ? 8 : 0);
      outg[(size_t)m * HH + ncol0 + l15] = acc[mt][v] + bv;
    }
  }
}

// ---------------------------------------------------------------------------
// Pass 2: sequential LSTM scan. One persistent workgroup of 1024 threads
// (32 waves). Wave w owns tile (mTile = w>>4, nTile = w&15) of the [32x256]
// state. u_i in LDS (one TDM load), h via LDS bf16, c in registers.
// ---------------------------------------------------------------------------
#define HPAD 264  // row pad (528 B, 16B-aligned) to spread LDS banks

__global__ __launch_bounds__(1024)
void vlstm_scan(const float* __restrict__ pres,
                const __bf16* __restrict__ uT,   // u_i transposed [N][K] bf16
                float* __restrict__ h_out,
                float* __restrict__ c_out,
                float* __restrict__ hseq) {
  __shared__ __bf16 uLDS[HH * HH];     // 128 KB: u_i^T, [n][k]
  __shared__ __bf16 hBF[BB][HPAD];     // ~17 KB

  const int tid   = threadIdx.x;
  const int lane  = tid & 31;
  const int wv    = tid >> 5;        // 0..31
  const int mTile = wv >> 4;         // 0..1  (batch rows 0-15 / 16-31)
  const int nTile = wv & 15;         // 0..15 (hidden cols)
  const int l15   = lane & 15;
  const int lhalf = lane >> 4;
  const int ncol  = nTile * 16 + l15;

  // u_i^T -> LDS with a single TDM op (256 rows x 256 bf16, contiguous).
  if (wv == 0) {
    tdm_load_2d_bf16(lds_off(&uLDS[0]), uT, 256, 256, HH, 256);
  }
  // h0 = 0
  {
    unsigned short* p = (unsigned short*)&hBF[0][0];
    for (int i = tid; i < BB * HPAD; i += 1024) p[i] = 0;
  }
  if (wv == 0) __builtin_amdgcn_s_wait_tensorcnt(0);
  __syncthreads();   // uLDS + hBF visible to all waves

  float c[8];
  float hv[8];
#pragma unroll
  for (int v = 0; v < 8; ++v) { c[v] = 0.0f; hv[v] = 0.0f; }

  const int arow = mTile * 16 + l15;              // A row (both lane halves)
  const __bf16* brow = &uLDS[(size_t)ncol * HH];  // this lane's u_i^T row

  for (int t = 0; t < TT; ++t) {
    // Hoist the gate pre-activation loads: issue before the WMMA chain so
    // their global latency hides under the matrix work.
    float pi[8], pf[8], pg[8], po[8];
#pragma unroll
    for (int v = 0; v < 8; ++v) {
      const int m = mTile * 16 + v + (lhalf ? 8 : 0);   // batch index
      const size_t base = ((size_t)m * TT + t) * HH + ncol;
      pi[v] = pres[base];
      pf[v] = pres[GSZ + base];
      pg[v] = pres[2 * GSZ + base];
      po[v] = pres[3 * GSZ + base];
    }

    // r = h @ u_i for this wave's 16x16 tile: 8 WMMAs, operands from LDS,
    // fragment loads pipelined one step deep.
    v8f r = {};
    v16bf afr, bfr;
    {
      const __bf16* ap = &hBF[arow][(lhalf ? 8 : 0)];
      afr = cat16(*(const v8bf*)ap, *(const v8bf*)(ap + 16));
      const __bf16* bp = brow + (lhalf ? 8 : 0);
      bfr = cat16(*(const v8bf*)bp, *(const v8bf*)(bp + 16));
    }
#pragma unroll
    for (int s = 0; s < 8; ++s) {
      v16bf anext, bnext;
      if (s + 1 < 8) {
        const int k0 = (s + 1) * 32 + (lhalf ? 8 : 0);
        const __bf16* ap = &hBF[arow][k0];
        anext = cat16(*(const v8bf*)ap, *(const v8bf*)(ap + 16));
        const __bf16* bp = brow + (s + 1) * 32 + (lhalf ? 8 : 0);
        bnext = cat16(*(const v8bf*)bp, *(const v8bf*)(bp + 16));
      }
      r = __builtin_amdgcn_wmma_f32_16x16x32_bf16(
          false, afr, false, bfr, (short)0, r, false, false);
      afr = anext;
      bfr = bnext;
    }

    // Gates + state update for the 8 (m, ncol) elements this lane owns.
#pragma unroll
    for (int v = 0; v < 8; ++v) {
      const int m = mTile * 16 + v + (lhalf ? 8 : 0);
      const size_t base = ((size_t)m * TT + t) * HH + ncol;
      const float rv = r[v];
      const float i = sigm(pi[v] + rv);
      const float f = sigm(pf[v] + rv);
      const float g = sigm(pg[v] + rv);
      const float o = sigm(po[v] + rv);
      const float cn = f + c[v] + i * g;
      const float hn = o + tanhf(cn);
      c[v] = cn;
      hv[v] = hn;
      hseq[base] = hn;  // hidden_seq[b, t, :]
    }

    __syncthreads();  // all waves done reading hBF for step t
#pragma unroll
    for (int v = 0; v < 8; ++v) {
      const int m = mTile * 16 + v + (lhalf ? 8 : 0);
      hBF[m][ncol] = (__bf16)hv[v];
    }
    __syncthreads();  // new h visible for step t+1
  }

  // Final h_t, c_t.
#pragma unroll
  for (int v = 0; v < 8; ++v) {
    const int m = mTile * 16 + v + (lhalf ? 8 : 0);
    h_out[(size_t)m * HH + ncol] = hv[v];
    c_out[(size_t)m * HH + ncol] = c[v];
  }
}

// ---------------------------------------------------------------------------
extern "C" void kernel_launch(void* const* d_in, const int* in_sizes, int n_in,
                              void* d_out, int out_size, void* d_ws, size_t ws_size,
                              hipStream_t stream) {
  (void)in_sizes; (void)n_in; (void)out_size; (void)ws_size;

  const float* x      = (const float*)d_in[0];
  const float* h_prev = (const float*)d_in[1];
  const float* w_i_x  = (const float*)d_in[2];
  const float* w_i_h  = (const float*)d_in[3];
  const float* w_f_x  = (const float*)d_in[4];
  const float* w_f_h  = (const float*)d_in[5];
  const float* w_c_x  = (const float*)d_in[6];
  const float* w_c_h  = (const float*)d_in[7];
  const float* w_o_x  = (const float*)d_in[8];
  const float* w_o_h  = (const float*)d_in[9];
  const float* u_i    = (const float*)d_in[10];
  const float* b_i    = (const float*)d_in[11];

  // Workspace layout: pres f32 [4*GSZ] | xbf | hbf | wT[8] | uT
  float*  pres = (float*)d_ws;
  __bf16* xbf  = (__bf16*)(pres + 4 * GSZ);
  __bf16* hbf  = xbf + GSZ;
  __bf16* wT   = hbf + GSZ;            // 8 * 256*256 bf16
  __bf16* uT   = wT + 8 * WSZ;         // 256*256 bf16

  float* out  = (float*)d_out;
  float* h_t  = out;                   // [32, 256]
  float* c_t  = out + BB * HH;         // [32, 256]
  float* hseq = out + 2 * BB * HH;     // [32, 2048, 256]

  vlstm_cvt_bf16<<<GSZ / (8 * 256), 256, 0, stream>>>(x, xbf);
  vlstm_cvt_bf16<<<GSZ / (8 * 256), 256, 0, stream>>>(h_prev, hbf);
  vlstm_cvt_wT<<<dim3(WSZ / 256, 1, 9), 256, 0, stream>>>(
      w_i_x, w_i_h, w_f_x, w_f_h, w_c_x, w_c_h, w_o_x, w_o_h, u_i, wT);

  vlstm_pre_gemm<<<dim3(BT / 256, 2, 4), 256, 0, stream>>>(
      xbf, hbf, wT, b_i, pres);

  vlstm_scan<<<1, 1024, 0, stream>>>(pres, uT, h_t, c_t, hseq);
}
